// LocalAttentionPathway_60713657697111
// MI455X (gfx1250) — matrix operand order repaired and checked by reading, not compiled, not run on hardware
//
#include <hip/hip_runtime.h>
#include <hip/hip_bf16.h>

typedef __bf16 bf16_t;
typedef __attribute__((ext_vector_type(16))) __bf16 v16bf;
typedef __attribute__((ext_vector_type(8)))  __bf16 v8bf;
typedef __attribute__((ext_vector_type(8)))  float  v8f;

#define NB    2
#define SEQ   2048
#define HDIM  1024
#define NHEAD 16
#define HD    64
#define WINSZ 256
#define NTOK  (NB * SEQ)   // 4096

// ---------------- WMMA fragment helpers (CDNA5 wave32 layouts) ----------------

// A-matrix 16x32 bf16: lane L holds row M=L%16.
// lanes 0-15:  elems 0..7 = K 0..7,  elems 8..15 = K 16..23
// lanes 16-31: elems 0..7 = K 8..15, elems 8..15 = K 24..31
__device__ __forceinline__ v16bf load_fragA(const bf16_t* __restrict__ rowptr, int half) {
  const v8bf lo = *(const v8bf*)(rowptr + half * 8);
  const v8bf hi = *(const v8bf*)(rowptr + 16 + half * 8);
  return __builtin_shufflevector(lo, hi, 0,1,2,3,4,5,6,7,8,9,10,11,12,13,14,15);
}

// B-matrix 32x16 bf16: lane L holds column N=L%16.
// lanes 0-15:  elems 0..7 = K 0..7,  elems 8..15 = K 16..23
// lanes 16-31: elems 0..7 = K 8..15, elems 8..15 = K 24..31
__device__ __forceinline__ v16bf load_fragB(const bf16_t* __restrict__ colptr, int half) {
  const v8bf lo = *(const v8bf*)(colptr + half * 8);
  const v8bf hi = *(const v8bf*)(colptr + 16 + half * 8);
  return __builtin_shufflevector(lo, hi, 0,1,2,3,4,5,6,7,8,9,10,11,12,13,14,15);
}

__device__ __forceinline__ v8f wmma_bf16(v16bf a, v16bf b, v8f c) {
  return __builtin_amdgcn_wmma_f32_16x16x32_bf16(false, a, false, b, (short)0, c, false, false);
}

// ---------------- fp32 -> bf16 conversion ----------------

__global__ void __launch_bounds__(256) cvt_kernel(const float* __restrict__ s,
                                                  bf16_t* __restrict__ d, int n) {
  int i = blockIdx.x * blockDim.x + threadIdx.x;
  if (i < n) d[i] = (bf16_t)s[i];
}

// ---------------- QKV projection + QK layernorm ----------------
// grid: (NTOK/16, 3)  block: 256 (8 waves). Wave w computes 8 col-tiles = dims [128w,128w+128) = heads 2w,2w+1.
__global__ void __launch_bounds__(256) qkv_ln_kernel(
    const bf16_t* __restrict__ xbf, const bf16_t* __restrict__ wbf,
    const float* __restrict__ bq, const float* __restrict__ bk, const float* __restrict__ bv,
    const float* __restrict__ lnqw, const float* __restrict__ lnqb,
    const float* __restrict__ lnkw, const float* __restrict__ lnkb,
    bf16_t* __restrict__ qout, bf16_t* __restrict__ kout, bf16_t* __restrict__ vTout)
{
  const int which = blockIdx.y;              // 0=q,1=k,2=v
  const int r0    = blockIdx.x * 16;         // token tile base
  const int wv    = threadIdx.x >> 5;
  const int lane  = threadIdx.x & 31;
  const int half  = lane >> 4;
  const int ln16  = lane & 15;
  const int c0    = wv * 128;

  const bf16_t* W    = wbf + (size_t)which * HDIM * HDIM;
  const float*  bias = (which == 0) ? bq : ((which == 1) ? bk : bv);

  v8f acc[8];
  #pragma unroll
  for (int t = 0; t < 8; t++)
    #pragma unroll
    for (int r = 0; r < 8; r++) acc[t][r] = 0.0f;

  const bf16_t* arow = xbf + (size_t)(r0 + ln16) * HDIM;
  for (int k = 0; k < HDIM; k += 32) {
    v16bf a = load_fragA(arow + k, half);
    #pragma unroll
    for (int t = 0; t < 8; t++) {
      const bf16_t* col = W + (size_t)(c0 + t * 16 + ln16) * HDIM + k;
      v16bf bfrg = load_fragB(col, half);
      acc[t] = wmma_bf16(a, bfrg, acc[t]);
    }
  }

  #pragma unroll
  for (int t = 0; t < 8; t++) {
    float bvv = bias[c0 + t * 16 + ln16];
    #pragma unroll
    for (int r = 0; r < 8; r++) acc[t][r] += bvv;
  }

  if (which < 2) {
    const float* gw = (which == 0) ? lnqw : lnkw;
    const float* gb = (which == 0) ? lnqb : lnkb;
    bf16_t* outp    = (which == 0) ? qout : kout;
    #pragma unroll
    for (int g = 0; g < 2; g++) {       // two heads per wave
      float sm[8], sq[8];
      #pragma unroll
      for (int r = 0; r < 8; r++) { sm[r] = 0.f; sq[r] = 0.f; }
      #pragma unroll
      for (int tt = 0; tt < 4; tt++) {
        int t = g * 4 + tt;
        #pragma unroll
        for (int r = 0; r < 8; r++) { float xv = acc[t][r]; sm[r] += xv; sq[r] += xv * xv; }
      }
      #pragma unroll
      for (int r = 0; r < 8; r++) {     // reduce over 16 lanes (N) within each half (rows differ by half)
        sm[r] += __shfl_xor(sm[r], 1, 32);  sq[r] += __shfl_xor(sq[r], 1, 32);
        sm[r] += __shfl_xor(sm[r], 2, 32);  sq[r] += __shfl_xor(sq[r], 2, 32);
        sm[r] += __shfl_xor(sm[r], 4, 32);  sq[r] += __shfl_xor(sq[r], 4, 32);
        sm[r] += __shfl_xor(sm[r], 8, 32);  sq[r] += __shfl_xor(sq[r], 8, 32);
        float mean = sm[r] * (1.0f / 64.0f);
        float var  = sq[r] * (1.0f / 64.0f) - mean * mean;
        sm[r] = mean;
        sq[r] = rsqrtf(var + 1e-5f);
      }
      #pragma unroll
      for (int tt = 0; tt < 4; tt++) {
        int t = g * 4 + tt;
        float gwv = gw[tt * 16 + ln16];
        float gbv = gb[tt * 16 + ln16];
        #pragma unroll
        for (int r = 0; r < 8; r++)
          acc[t][r] = (acc[t][r] - sm[r]) * sq[r] * gwv + gbv;
      }
    }
    #pragma unroll
    for (int t = 0; t < 8; t++) {
      int col = c0 + t * 16 + ln16;
      #pragma unroll
      for (int r = 0; r < 8; r++) {
        int row = r0 + r + 8 * half;
        outp[(size_t)row * HDIM + col] = (bf16_t)acc[t][r];
      }
    }
  } else {
    // V: store transposed per (b,h): vT[((b*NH+h)*HD+d)*SEQ + s]
    #pragma unroll
    for (int t = 0; t < 8; t++) {
      int dim = c0 + t * 16 + ln16;
      int h = dim >> 6, d = dim & 63;
      #pragma unroll
      for (int r = 0; r < 8; r++) {
        int tok = r0 + r + 8 * half;
        int bb = tok >> 11, sp = tok & (SEQ - 1);
        vTout[((size_t)((bb * NHEAD + h) * HD + d)) * SEQ + sp] = (bf16_t)acc[t][r];
      }
    }
  }
}

// ---------------- Windowed causal attention ----------------
// grid: (SEQ/16, NHEAD, NB), block: 32 (one wave per 16-query tile).
__global__ void __launch_bounds__(32) attn_kernel(
    const bf16_t* __restrict__ qbf, const bf16_t* __restrict__ kbf,
    const bf16_t* __restrict__ vT, bf16_t* __restrict__ aout)
{
  __shared__ float sc[16][292];
  __shared__ float rinv[16];

  const int qt = blockIdx.x, h = blockIdx.y, b = blockIdx.z;
  const int lane = threadIdx.x & 31;
  const int half = lane >> 4, ln16 = lane & 15;
  const int m0   = qt * 16;
  const int jlo  = (m0 >= WINSZ) ? (m0 - WINSZ) : 0;
  const int ncols = m0 + 16 - jlo;         // <= 272
  const int ntile = ncols >> 4;            // <= 17
  const size_t rowbase = (size_t)b * SEQ + m0;

  for (int idx = lane; idx < 16 * 292; idx += 32) ((float*)sc)[idx] = 0.0f;
  __syncthreads();

  const bf16_t* qrow = qbf + (rowbase + ln16) * (size_t)HDIM + h * HD;
  v16bf aq0 = load_fragA(qrow, half);
  v16bf aq1 = load_fragA(qrow + 32, half);

  for (int t = 0; t < ntile; t++) {
    int jbase = jlo + t * 16;
    const bf16_t* krow = kbf + ((size_t)b * SEQ + jbase + ln16) * HDIM + h * HD;
    v16bf bk0 = load_fragB(krow, half);
    v16bf bk1 = load_fragB(krow + 32, half);
    v8f s;
    #pragma unroll
    for (int r = 0; r < 8; r++) s[r] = 0.0f;
    s = wmma_bf16(aq0, bk0, s);
    s = wmma_bf16(aq1, bk1, s);
    int j = jbase + ln16;
    #pragma unroll
    for (int r = 0; r < 8; r++) {
      int m = r + 8 * half;
      int i = m0 + m;
      float val = s[r] * 0.125f;                      // 1/sqrt(64)
      bool ok = (j <= i) && (i - j <= WINSZ);
      sc[m][t * 16 + ln16] = ok ? val : -1e30f;
    }
  }
  __syncthreads();

  for (int m = 0; m < 16; m++) {
    float mx = -1e30f;
    for (int c = lane; c < ncols; c += 32) mx = fmaxf(mx, sc[m][c]);
    mx = fmaxf(mx, __shfl_xor(mx, 16, 32));
    mx = fmaxf(mx, __shfl_xor(mx, 8, 32));
    mx = fmaxf(mx, __shfl_xor(mx, 4, 32));
    mx = fmaxf(mx, __shfl_xor(mx, 2, 32));
    mx = fmaxf(mx, __shfl_xor(mx, 1, 32));
    float sum = 0.f;
    for (int c = lane; c < ncols; c += 32) {
      float e = __expf(sc[m][c] - mx);
      sc[m][c] = e;
      sum += e;
    }
    sum += __shfl_xor(sum, 16, 32);
    sum += __shfl_xor(sum, 8, 32);
    sum += __shfl_xor(sum, 4, 32);
    sum += __shfl_xor(sum, 2, 32);
    sum += __shfl_xor(sum, 1, 32);
    if (lane == 0) rinv[m] = 1.0f / sum;
  }
  __syncthreads();

  const int nch = (ncols + 31) >> 5;       // <= 9
  v8f o[4];
  #pragma unroll
  for (int dt = 0; dt < 4; dt++)
    #pragma unroll
    for (int r = 0; r < 8; r++) o[dt][r] = 0.0f;

  for (int ch = 0; ch < nch; ch++) {
    int cb = ch * 32;
    v16bf ap;                               // P in A layout from LDS (row = ln16)
    int base1 = cb + half * 16;
    int base2 = cb + half * 16 + 8;
    #pragma unroll
    for (int e = 0; e < 8; e++) {
      ap[e]     = (bf16_t)sc[ln16][base1 + e];
      ap[8 + e] = (bf16_t)sc[ln16][base2 + e];
    }
    int jb = jlo + cb + half * 16;
    #pragma unroll
    for (int dt = 0; dt < 4; dt++) {
      const bf16_t* vrow = vT + ((size_t)((b * NHEAD + h) * HD + dt * 16 + ln16)) * SEQ;
      v16bf bvv;
      if (jb + 16 <= SEQ) {
        const v8bf lo = *(const v8bf*)(vrow + jb);
        const v8bf hi = *(const v8bf*)(vrow + jb + 8);
        bvv = __builtin_shufflevector(lo, hi, 0,1,2,3,4,5,6,7,8,9,10,11,12,13,14,15);
      } else {
        #pragma unroll
        for (int e = 0; e < 16; e++) {      // P is zero past ncols; clamp just keeps reads in-bounds
          int j = jb + e;
          if (j >= SEQ) j = SEQ - 1;
          bvv[e] = vrow[j];
        }
      }
      o[dt] = wmma_bf16(ap, bvv, o[dt]);
    }
  }

  #pragma unroll
  for (int dt = 0; dt < 4; dt++) {
    #pragma unroll
    for (int r = 0; r < 8; r++) {
      int m = r + 8 * half;
      float val = o[dt][r] * rinv[m];
      aout[(rowbase + m) * (size_t)HDIM + h * HD + dt * 16 + ln16] = (bf16_t)val;
    }
  }
}

// ---------------- Output projection (fp32 out) ----------------
__global__ void __launch_bounds__(256) oproj_kernel(
    const bf16_t* __restrict__ abf, const bf16_t* __restrict__ Wo,
    const float* __restrict__ bo, float* __restrict__ out)
{
  const int r0   = blockIdx.x * 16;
  const int wv   = threadIdx.x >> 5;
  const int lane = threadIdx.x & 31;
  const int half = lane >> 4, ln16 = lane & 15;
  const int c0   = wv * 128;

  v8f acc[8];
  #pragma unroll
  for (int t = 0; t < 8; t++)
    #pragma unroll
    for (int r = 0; r < 8; r++) acc[t][r] = 0.0f;

  const bf16_t* arow = abf + (size_t)(r0 + ln16) * HDIM;
  for (int k = 0; k < HDIM; k += 32) {
    v16bf a = load_fragA(arow + k, half);
    #pragma unroll
    for (int t = 0; t < 8; t++) {
      const bf16_t* col = Wo + (size_t)(c0 + t * 16 + ln16) * HDIM + k;
      v16bf bfrg = load_fragB(col, half);
      acc[t] = wmma_bf16(a, bfrg, acc[t]);
    }
  }

  #pragma unroll
  for (int t = 0; t < 8; t++) {
    int col = c0 + t * 16 + ln16;
    float bvv = bo[col];
    #pragma unroll
    for (int r = 0; r < 8; r++) {
      int row = r0 + r + 8 * half;
      out[(size_t)row * HDIM + col] = acc[t][r] + bvv;
    }
  }
}

// ---------------- launch ----------------

extern "C" void kernel_launch(void* const* d_in, const int* in_sizes, int n_in,
                              void* d_out, int out_size, void* d_ws, size_t ws_size,
                              hipStream_t stream) {
  const float* x    = (const float*)d_in[0];
  const float* Wq   = (const float*)d_in[1];
  const float* bq   = (const float*)d_in[2];
  const float* Wk   = (const float*)d_in[3];
  const float* bk   = (const float*)d_in[4];
  const float* Wv   = (const float*)d_in[5];
  const float* bv   = (const float*)d_in[6];
  const float* Wo   = (const float*)d_in[7];
  const float* bo   = (const float*)d_in[8];
  const float* lnqw = (const float*)d_in[9];
  const float* lnqb = (const float*)d_in[10];
  const float* lnkw = (const float*)d_in[11];
  const float* lnkb = (const float*)d_in[12];
  float* out = (float*)d_out;

  char* ws = (char*)d_ws;
  const size_t xsz = (size_t)NTOK * HDIM;       // 4 Mi elements
  const size_t wsz = (size_t)HDIM * HDIM;       // 1 Mi elements
  bf16_t* xbf  = (bf16_t*)ws;  ws += xsz * sizeof(bf16_t);
  bf16_t* wbf  = (bf16_t*)ws;  ws += 4 * wsz * sizeof(bf16_t);
  bf16_t* qbf  = (bf16_t*)ws;  ws += xsz * sizeof(bf16_t);
  bf16_t* kbf  = (bf16_t*)ws;  ws += xsz * sizeof(bf16_t);
  bf16_t* vTbf = (bf16_t*)ws;  ws += xsz * sizeof(bf16_t);
  bf16_t* abf  = (bf16_t*)ws;  ws += xsz * sizeof(bf16_t);
  (void)ws_size; (void)in_sizes; (void)n_in; (void)out_size;

  cvt_kernel<<<(int)(xsz / 256), 256, 0, stream>>>(x,  xbf, (int)xsz);
  cvt_kernel<<<(int)(wsz / 256), 256, 0, stream>>>(Wq, wbf + 0 * wsz, (int)wsz);
  cvt_kernel<<<(int)(wsz / 256), 256, 0, stream>>>(Wk, wbf + 1 * wsz, (int)wsz);
  cvt_kernel<<<(int)(wsz / 256), 256, 0, stream>>>(Wv, wbf + 2 * wsz, (int)wsz);
  cvt_kernel<<<(int)(wsz / 256), 256, 0, stream>>>(Wo, wbf + 3 * wsz, (int)wsz);

  qkv_ln_kernel<<<dim3(NTOK / 16, 3), 256, 0, stream>>>(
      xbf, wbf, bq, bk, bv, lnqw, lnqb, lnkw, lnkb, qbf, kbf, vTbf);

  attn_kernel<<<dim3(SEQ / 16, NHEAD, NB), 32, 0, stream>>>(qbf, kbf, vTbf, abf);

  oproj_kernel<<<NTOK / 16, 256, 0, stream>>>(abf, wbf + 3 * wsz, bo, out);
}
